// ETFaithfulGraphModel_32538672234679
// MI455X (gfx1250) — matrix operand order repaired and checked by reading, not compiled, not run on hardware
//
#include <hip/hip_runtime.h>
#include <hip/hip_bf16.h>
#include <cstdint>

// Problem constants (match reference)
#define Dd   256
#define Hh   8
#define Zz   64
#define Bb   4
#define Nn   2048
#define HZ   512   // Hh*Zz

typedef __attribute__((ext_vector_type(16))) _Float16 v16h;
typedef __attribute__((ext_vector_type(8)))  _Float16 half8;
typedef __attribute__((ext_vector_type(4)))  _Float16 half4;
typedef __attribute__((ext_vector_type(8)))  float    v8f;
typedef __attribute__((ext_vector_type(2)))  float    f32x2;

// ---------------------------------------------------------------------------
// Kernel 0: zero the scalar output (harness poisons d_out with 0xAA)
// ---------------------------------------------------------------------------
__global__ void et_zero(float* out) {
    if (threadIdx.x == 0) out[0] = 0.0f;
}

// ---------------------------------------------------------------------------
// Kernel 1: bulk f32 -> f16 conversion (g, Wq, Wk), done once.
// ---------------------------------------------------------------------------
__global__ __launch_bounds__(256)
void et_cvt(const float* __restrict__ src, _Float16* __restrict__ dst, int n) {
    const int stride = gridDim.x * blockDim.x * 4;
    for (int i = (blockIdx.x * blockDim.x + threadIdx.x) * 4; i < n; i += stride) {
        float4 v = *(const float4*)(src + i);
        half4 h;
        h[0] = (_Float16)v.x; h[1] = (_Float16)v.y;
        h[2] = (_Float16)v.z; h[3] = (_Float16)v.w;
        *(half4*)(dst + i) = h;
    }
}

// ---------------------------------------------------------------------------
// Kernel 2: q/k projection.  q[b,n,h,z] = sum_d g[b,n,d] * W[h,z,d]
// GEMM  C[32 x 512] = G[32 x 256] x W^T[256 x 512]   per (b, 32-row n-tile).
// Each W (B) fragment is reused by two A fragments (rows 0-15 / 16-31).
// Output layout: q16/k16[b][n][h*64+z]  (f16)
// ---------------------------------------------------------------------------
__global__ __launch_bounds__(128)
void et_project(const _Float16* __restrict__ g16,
                const _Float16* __restrict__ w16q,
                const _Float16* __restrict__ w16k,
                _Float16* __restrict__ q16,
                _Float16* __restrict__ k16) {
    const int b  = blockIdx.y;
    const int n0 = blockIdx.x * 32;
    const int t  = threadIdx.x;
    const int w  = t >> 5;        // wave id (wave32)
    const int L  = t & 31;        // lane id
    const int r  = L & 15;
    const int hb = L >> 4;        // half-wave selector

    // g tile (32 x 256 f16); +8 half pad per row -> 4-bank row skew
    __shared__ _Float16 gsh[32][Dd + 8];

    // async-stage g tile: 32 rows x 32 granules of 16B
    const _Float16* gtile = g16 + ((size_t)b * Nn + n0) * Dd;
    for (int i = t; i < 32 * (Dd / 8); i += 128) {
        int row = i >> 5, c = i & 31;
        unsigned goff = (unsigned)((row * Dd + c * 8) * sizeof(_Float16));
        auto lp = (__attribute__((address_space(3))) _Float16*)&gsh[row][c * 8];
        asm volatile("global_load_async_to_lds_b128 %0, %1, %2 offset:0"
                     :: "v"(lp), "v"(goff), "s"(gtile) : "memory");
    }
    asm volatile("s_wait_asynccnt 0" ::: "memory");
    __syncthreads();

    // each wave owns hz-tiles  w, w+4, ...  (32 tiles of 16 columns)
    for (int tile = w; tile < HZ / 16; tile += 4) {
#pragma unroll
        for (int which = 0; which < 2; ++which) {
            const _Float16* W = which ? w16k : w16q;
            _Float16* outb = (which ? k16 : q16) + ((size_t)b * Nn + n0) * HZ;
            v8f acc0 = {}, acc1 = {};
#pragma unroll
            for (int dc = 0; dc < Dd / 32; ++dc) {
                // B fragment (32x16): lane holds column (hz row of W), K = hb*8 + i, 16 + hb*8 + i
                const _Float16* wp = W + (size_t)(tile * 16 + r) * Dd + dc * 32 + hb * 8;
                half8 b0 = *(const half8*)wp;
                half8 b1 = *(const half8*)(wp + 16);
                v16h bf;
#pragma unroll
                for (int i = 0; i < 8; ++i) { bf[i] = b0[i]; bf[8 + i] = b1[i]; }
                // A fragments (16x32 f16) for the two 16-row sub-tiles
                half8 a0 = *(const half8*)&gsh[r][dc * 32 + hb * 8];
                half8 a1 = *(const half8*)&gsh[r][dc * 32 + 16 + hb * 8];
                v16h a;
#pragma unroll
                for (int i = 0; i < 8; ++i) { a[i] = a0[i]; a[8 + i] = a1[i]; }
                acc0 = __builtin_amdgcn_wmma_f32_16x16x32_f16(
                        false, a, false, bf, (short)0, acc0, false, false);
                half8 a2 = *(const half8*)&gsh[16 + r][dc * 32 + hb * 8];
                half8 a3 = *(const half8*)&gsh[16 + r][dc * 32 + 16 + hb * 8];
                v16h a_;
#pragma unroll
                for (int i = 0; i < 8; ++i) { a_[i] = a2[i]; a_[8 + i] = a3[i]; }
                acc1 = __builtin_amdgcn_wmma_f32_16x16x32_f16(
                        false, a_, false, bf, (short)0, acc1, false, false);
            }
            // C/D layout: VGPR v -> row v + hb*8, lane col r
            const int col = tile * 16 + r;
#pragma unroll
            for (int vv = 0; vv < 8; ++vv) {
                outb[(size_t)(vv + hb * 8) * HZ + col]      = (_Float16)acc0[vv];
                outb[(size_t)(16 + vv + hb * 8) * HZ + col] = (_Float16)acc1[vv];
            }
        }
    }
}

// ---------------------------------------------------------------------------
// Kernel 3: fused scores + head-mix + masked sum-exp + energy.
// Workgroup = 32 query rows, 2 waves; wave w owns rows [16w, 16w+16).
// Both waves march over key tiles in lockstep; the 16x512 f16 k tile is
// async-staged into a double-buffered LDS tile shared by both waves.
// Head-mix forced onto V_PK_FMA_F32 (packed 2xf32 VOP3P).
// ---------------------------------------------------------------------------
__global__ __launch_bounds__(64)
void et_energy(const _Float16* __restrict__ q16,
               const _Float16* __restrict__ k16,
               const float* __restrict__ Hw,
               const float* __restrict__ betas,
               const int*  __restrict__ dense_sizes,
               float* __restrict__ out) {
    const int b    = blockIdx.y;
    const int n0   = blockIdx.x * 32;
    const int t    = threadIdx.x;
    const int w    = t >> 5;
    const int L    = t & 31;
    const int r    = L & 15;
    const int hb   = L >> 4;
    const int size = dense_sizes[b];

    __shared__ _Float16 qsh[32][HZ + 8];       // q tile (both sub-tiles)
    __shared__ _Float16 kbuf[2][16][HZ + 8];   // double-buffered k tile
    __shared__ float    bh[Hh * Hh];           // betas[h] * Hw[h][g]
    __shared__ float    sacc[32][Hh];          // per (n_local, g) sum of exp
    __shared__ float    eacc;                  // block partial energy

    if (t == 0) eacc = 0.0f;
    bh[t] = betas[t >> 3] * Hw[t];             // 64 threads == 64 entries

    if (n0 >= size) return;                    // fully masked rows

    // ---- prefetch q tile (32 x 512 f16) and k tile 0 (async) --------------
    const _Float16* qtile = q16 + ((size_t)b * Nn + n0) * HZ;
    for (int i = t; i < 32 * (HZ / 8); i += 64) {     // 2048 x 16B granules
        int row = i >> 6, c = i & 63;
        unsigned goff = (unsigned)((row * HZ + c * 8) * sizeof(_Float16));
        auto lp = (__attribute__((address_space(3))) _Float16*)&qsh[row][c * 8];
        asm volatile("global_load_async_to_lds_b128 %0, %1, %2 offset:0"
                     :: "v"(lp), "v"(goff), "s"(qtile) : "memory");
    }
    const _Float16* kbase = k16 + (size_t)b * Nn * HZ;
    {
        const _Float16* ktile = kbase;                // tile 0
        for (int i = t; i < 16 * (HZ / 8); i += 64) { // 1024 x 16B granules
            int row = i >> 6, c = i & 63;
            unsigned goff = (unsigned)((row * HZ + c * 8) * sizeof(_Float16));
            auto lp = (__attribute__((address_space(3))) _Float16*)&kbuf[0][row][c * 8];
            asm volatile("global_load_async_to_lds_b128 %0, %1, %2 offset:0"
                         :: "v"(lp), "v"(goff), "s"(ktile) : "memory");
        }
    }

    const int mtiles = (size + 15) >> 4;

    float acc[8][8];                           // sum-exp accum: [row-vgpr][g]
#pragma unroll
    for (int vv = 0; vv < 8; ++vv)
#pragma unroll
        for (int gg = 0; gg < 8; ++gg) acc[vv][gg] = 0.0f;

    const int qrow = w * 16 + r;               // this wave's query row in LDS

    for (int mt = 0; mt < mtiles; ++mt) {
        const int cur = mt & 1;
        // wait for tile mt (and, on mt==0, the q tile) to land, then sync
        asm volatile("s_wait_asynccnt 0" ::: "memory");
        __syncthreads();
        // prefetch tile mt+1 into the other buffer (overlaps compute below);
        // safe: that buffer was last read in iteration mt-1, all waves have
        // passed the barrier above since then.
        if (mt + 1 < mtiles) {
            const _Float16* ktile = kbase + (size_t)(mt + 1) * 16 * HZ;
            const int nxt = cur ^ 1;
            for (int i = t; i < 16 * (HZ / 8); i += 64) {
                int row = i >> 6, c = i & 63;
                unsigned goff = (unsigned)((row * HZ + c * 8) * sizeof(_Float16));
                auto lp = (__attribute__((address_space(3))) _Float16*)&kbuf[nxt][row][c * 8];
                asm volatile("global_load_async_to_lds_b128 %0, %1, %2 offset:0"
                             :: "v"(lp), "v"(goff), "s"(ktile) : "memory");
            }
        }

        f32x2 lg2[8][4];                       // packed logits: [row-vgpr][g-pair]
#pragma unroll
        for (int vv = 0; vv < 8; ++vv)
#pragma unroll
            for (int gp = 0; gp < 4; ++gp) lg2[vv][gp] = (f32x2){0.0f, 0.0f};

#pragma unroll
        for (int h = 0; h < Hh; ++h) {
            v8f c = {};
#pragma unroll
            for (int zc = 0; zc < 2; ++zc) {           // K = Z = 64 -> 2 x k32
                const int zb = h * 64 + zc * 32;
                half8 a0 = *(const half8*)&qsh[qrow][zb + hb * 8];
                half8 a1 = *(const half8*)&qsh[qrow][zb + 16 + hb * 8];
                v16h a;
#pragma unroll
                for (int i = 0; i < 8; ++i) { a[i] = a0[i]; a[8 + i] = a1[i]; }
                // B = k^T fragment: lane holds key row m = mt*16 + r,
                // z = zb + hb*8 + i and zb + 16 + hb*8 + i -> two 16B LDS loads
                const _Float16* kp = &kbuf[cur][r][zb + hb * 8];
                half8 b0 = *(const half8*)kp;
                half8 b1 = *(const half8*)(kp + 16);
                v16h bf;
#pragma unroll
                for (int i = 0; i < 8; ++i) { bf[i] = b0[i]; bf[8 + i] = b1[i]; }
                c = __builtin_amdgcn_wmma_f32_16x16x32_f16(
                        false, a, false, bf, (short)0, c, false, false);
            }
            // head mix: packed f32 FMA (v_pk_fma_f32), co-executes with WMMA
#pragma unroll
            for (int gp = 0; gp < 4; ++gp) {
                const f32x2 mhg = *(const f32x2*)&bh[h * 8 + gp * 2];  // ds_load_b64
#pragma unroll
                for (int vv = 0; vv < 8; ++vv) {
                    f32x2 cs = {c[vv], c[vv]};
                    asm("v_pk_fma_f32 %0, %1, %2, %0"
                        : "+v"(lg2[vv][gp])
                        : "v"(cs), "v"(mhg));
                }
            }
        }
        // masked exp-accumulate; lane's key index m = mt*16 + r
        if (mt * 16 + r < size) {
#pragma unroll
            for (int vv = 0; vv < 8; ++vv)
#pragma unroll
                for (int gg = 0; gg < 8; ++gg)
                    acc[vv][gg] += __expf(lg2[vv][gg >> 1][gg & 1]);
        }
        __syncthreads();   // all waves done reading kbuf[cur] before reuse
    }

    // ---- reduce over the 16 key lanes of each half-wave -------------------
    // Each wave owns its 16 rows exclusively -> plain LDS stores.
#pragma unroll
    for (int vv = 0; vv < 8; ++vv) {
#pragma unroll
        for (int gg = 0; gg < 8; ++gg) {
            float s = acc[vv][gg];
            s += __shfl_xor(s, 1, 32);
            s += __shfl_xor(s, 2, 32);
            s += __shfl_xor(s, 4, 32);
            s += __shfl_xor(s, 8, 32);         // stays within 16-lane group
            if (r == gg) sacc[w * 16 + vv + hb * 8][gg] = s;
        }
    }
    __syncthreads();

    // ---- lse -> energy partial (32 rows x 8 heads = 256 terms, 64 thr) ----
    float term = 0.0f;
#pragma unroll
    for (int j = 0; j < 4; ++j) {
        const int idx = t + 64 * j;
        const int nl = idx >> 3, gg = idx & 7;
        if (n0 + nl < size)
            term += -__logf(sacc[nl][gg]) / betas[gg];
    }
    term += __shfl_xor(term, 1, 32);
    term += __shfl_xor(term, 2, 32);
    term += __shfl_xor(term, 4, 32);
    term += __shfl_xor(term, 8, 32);
    term += __shfl_xor(term, 16, 32);
    if (L == 0) atomicAdd(&eacc, term);        // ds_add_f32
    __syncthreads();
    if (t == 0) atomicAdd(out, eacc);          // global_atomic_add_f32
}

// ---------------------------------------------------------------------------
extern "C" void kernel_launch(void* const* d_in, const int* in_sizes, int n_in,
                              void* d_out, int out_size, void* d_ws, size_t ws_size,
                              hipStream_t stream) {
    (void)in_sizes; (void)n_in; (void)out_size; (void)ws_size;
    const float* g     = (const float*)d_in[0];
    const float* Wq    = (const float*)d_in[1];
    const float* Wk    = (const float*)d_in[2];
    const float* Hw    = (const float*)d_in[3];
    const float* betas = (const float*)d_in[4];
    const int*   sizes = (const int*)d_in[5];
    float* out = (float*)d_out;

    // workspace layout (f16): g16 | w16q | w16k | q16 | k16
    _Float16* g16  = (_Float16*)d_ws;                          // 4 MB
    _Float16* w16q = g16  + (size_t)Bb * Nn * Dd;              // 256 KB
    _Float16* w16k = w16q + (size_t)Hh * Zz * Dd;              // 256 KB
    _Float16* q16  = w16k + (size_t)Hh * Zz * Dd;              // 8 MB
    _Float16* k16  = q16  + (size_t)Bb * Nn * HZ;              // 8 MB

    et_zero<<<1, 32, 0, stream>>>(out);
    et_cvt<<<512, 256, 0, stream>>>(g,  g16,  Bb * Nn * Dd);
    et_cvt<<<64,  256, 0, stream>>>(Wq, w16q, Hh * Zz * Dd);
    et_cvt<<<64,  256, 0, stream>>>(Wk, w16k, Hh * Zz * Dd);
    et_project<<<dim3(Nn / 32, Bb), 128, 0, stream>>>(g16, w16q, w16k, q16, k16);
    et_energy<<<dim3(Nn / 32, Bb), 64, 0, stream>>>(q16, k16, Hw, betas, sizes, out);
}
